// TranslationModel_5738076307586
// MI455X (gfx1250) — matrix operand order repaired and checked by reading, not compiled, not run on hardware
//
#include <hip/hip_runtime.h>

// ---------------------------------------------------------------------------
// Types for CDNA5 WMMA (wave32): V_WMMA_F32_16X16X32_BF16
// ---------------------------------------------------------------------------
typedef __attribute__((ext_vector_type(16))) __bf16        v16bf;
typedef __attribute__((ext_vector_type(8)))  float         v8f;
typedef __attribute__((ext_vector_type(4)))  unsigned int  v4u;

__device__ __forceinline__ int lane32() { return (int)(threadIdx.x & 31u); }

// A-fragment (16x32 bf16). 'row' = this lane's M-row, 32 contiguous bf16
// (16B aligned). Layout (ISA 7.12.2): lanes 0-15 -> K {0..7,16..23},
// lanes 16-31 -> +8. Two b128 loads; splat zero for conv padding.
__device__ __forceinline__ v16bf load_A_seg(const __bf16* __restrict__ row, bool valid) {
    const int kb = (lane32() >> 4) << 3;      // 0 or 8
    union { v4u q[2]; v16bf v; } u;
    if (valid) {
        u.q[0] = *(const v4u*)(row + kb);        // elements 0..7  : K kb..kb+7
        u.q[1] = *(const v4u*)(row + kb + 16);   // elements 8..15 : K kb+16..kb+23
    } else {
        u.q[0] = (v4u)0u;
        u.q[1] = (v4u)0u;
    }
    return u.v;
}

// B-fragment from PRE-SWIZZLED weights: fragment = 32 lanes x 16 bf16, stored
// so lane l's register image (element e <-> K = ((l>>4)<<3)+(e<8 ? e : e+8), N = l&15) is
// 32 contiguous bytes. Whole fragment = contiguous 1KB -> 2 coalesced b128.
__device__ __forceinline__ v16bf load_B_packed(const __bf16* __restrict__ frag) {
    const __bf16* p = frag + (size_t)lane32() * 16;
    union { v4u q[2]; v16bf v; } u;
    u.q[0] = *(const v4u*)(p);
    u.q[1] = *(const v4u*)(p + 8);
    return u.v;
}

__device__ __forceinline__ v8f wmma_bf16(v16bf a, v16bf b, v8f c) {
    return __builtin_amdgcn_wmma_f32_16x16x32_bf16(false, a, false, b, (short)0, c,
                                                   false, false);
}

// C/D layout: element r of v8f belongs to (m = r + 8*(lane>=16), n = lane&15)

// ---------------------------------------------------------------------------
// Weight repack: OIHW fp32 -> per-fragment lane-swizzled bf16.
//   wpk1: [2 frags]          (conv1, K = tap*3+ci padded 27->32)
//   wpk2: [9 taps][4 frags]  (conv2, K-chunk = tap, k = ci)
//   wpk3: [9][2 kk][8 frags] (conv3, k = kk*32 + ci_in_chunk)
// Fragment element: l = lane, e = 0..15 -> K = ((l>>4)<<3)+(e<8 ? e : e+8), N(co) = l&15.
// ---------------------------------------------------------------------------
__global__ void pack_weights_kernel(const float* __restrict__ w1,
                                    const float* __restrict__ w2,
                                    const float* __restrict__ w3,
                                    __bf16* __restrict__ wpk1,   // 2*512
                                    __bf16* __restrict__ wpk2,   // 36*512
                                    __bf16* __restrict__ wpk3) { // 144*512
    const int t = blockIdx.x * 256 + (int)threadIdx.x;
    if (t < 2 * 512) {
        const int frag = t >> 9, rem = t & 511, l = rem >> 4, e = rem & 15;
        const int k  = ((l >> 4) << 3) + (e < 8 ? e : e + 8); // 0..31
        const int co = frag * 16 + (l & 15);          // 0..31
        float v = 0.0f;
        if (k < 27) {
            const int tap = k / 3, ci = k % 3;
            v = w1[((co * 3 + ci) * 3 + tap / 3) * 3 + (tap % 3)];
        }
        wpk1[t] = (__bf16)v;
    }
    const int t2 = t - 2 * 512;
    if (t2 >= 0 && t2 < 36 * 512) {
        const int frag = t2 >> 9, rem = t2 & 511, l = rem >> 4, e = rem & 15;
        const int tap = frag >> 2, ct = frag & 3;
        const int ci = ((l >> 4) << 3) + (e < 8 ? e : e + 8); // 0..31
        const int co = ct * 16 + (l & 15);            // 0..63
        wpk2[t2] = (__bf16)w2[((co * 32 + ci) * 3 + tap / 3) * 3 + (tap % 3)];
    }
    const int t3 = t - 2 * 512 - 36 * 512;
    if (t3 >= 0 && t3 < 144 * 512) {
        const int frag = t3 >> 9, rem = t3 & 511, l = rem >> 4, e = rem & 15;
        const int tap = frag >> 4, kkt = frag & 15;
        const int kk = kkt >> 3, ct = kkt & 7;
        const int ci = kk * 32 + ((l >> 4) << 3) + (e < 8 ? e : e + 8); // 0..63
        const int co = ct * 16 + (l & 15);            // 0..127
        wpk3[t3] = (__bf16)w3[((co * 64 + ci) * 3 + tap / 3) * 3 + (tap % 3)];
    }
}

// ---------------------------------------------------------------------------
// conv1: 3->32, stride 1, pad 1. In: NCHW fp32 [B,3,256,256].
// Out: NHWC bf16 [B,256,256,32]. K (27, padded 32) = ONE wmma chunk.
// Per wave: 16 pixels x 32 channels (2 WMMA). Block = 8 waves = 128 px.
// ---------------------------------------------------------------------------
__global__ void conv1_wmma_kernel(const float* __restrict__ img,
                                  const __bf16* __restrict__ wpk1,
                                  const float* __restrict__ bias,
                                  __bf16* __restrict__ out) {
    const int b  = blockIdx.z;
    const int y  = blockIdx.y;
    const int w  = (int)threadIdx.x >> 5;
    const int x0 = blockIdx.x * 128 + w * 16;
    const int l  = lane32();
    const int m  = l & 15;
    const int x  = x0 + m;
    const int kb = (l >> 4) << 3;

    // Gather A fragment from NCHW fp32 image (k = tap*3 + ci).
    v16bf A;
#pragma unroll
    for (int i = 0; i < 16; ++i) {
        const int k = kb + ((i < 8) ? i : (8 + i));   // i>=8 -> kb+16+(i-8)
        float v = 0.0f;
        if (k < 27) {
            const int tap = k / 3, ci = k % 3;
            const int iy = y + tap / 3 - 1, ix = x + tap % 3 - 1;
            if (iy >= 0 && iy < 256 && ix >= 0 && ix < 256)
                v = img[(((size_t)b * 3 + ci) * 256 + iy) * 256 + ix];
        }
        A[i] = (__bf16)v;
    }

    v8f acc[2];
#pragma unroll
    for (int t = 0; t < 2; ++t) {
#pragma unroll
        for (int r = 0; r < 8; ++r) acc[t][r] = 0.0f;
        v16bf B = load_B_packed(wpk1 + t * 512);
        acc[t] = wmma_bf16(A, B, acc[t]);
    }

    const size_t rowbase = ((size_t)b * 256 + y) * 256;
#pragma unroll
    for (int t = 0; t < 2; ++t) {
        const int   co = t * 16 + (l & 15);
        const float bb = bias[co];
#pragma unroll
        for (int r = 0; r < 8; ++r) {
            const int mm = r + ((l >> 4) << 3);
            float v = acc[t][r] + bb;
            v = v > 0.0f ? v : 0.0f;
            out[(rowbase + x0 + mm) * 32 + co] = (__bf16)v;
        }
    }
}

// ---------------------------------------------------------------------------
// conv2: 32->64, stride 2, pad 1. NHWC bf16 [B,256,256,32] -> [B,128,128,64].
// Per wave: 16 px x 64 co; 9 taps x (1 A + 4 B-frag + 4 WMMA) = 36 WMMA.
// ---------------------------------------------------------------------------
__global__ void conv2_wmma_kernel(const __bf16* __restrict__ in,
                                  const __bf16* __restrict__ wpk2,
                                  const float* __restrict__ bias,
                                  __bf16* __restrict__ out) {
    const int b  = blockIdx.y;
    const int y  = blockIdx.x;                 // 0..127
    const int w  = (int)threadIdx.x >> 5;
    const int x0 = w * 16;
    const int l  = lane32();
    const int m  = l & 15;
    const int x  = x0 + m;

    v8f acc[4];
#pragma unroll
    for (int t = 0; t < 4; ++t)
#pragma unroll
        for (int r = 0; r < 8; ++r) acc[t][r] = 0.0f;

#pragma unroll
    for (int tap = 0; tap < 9; ++tap) {
        const int iy = 2 * y - 1 + tap / 3;
        const int ix = 2 * x - 1 + tap % 3;
        const bool valid = (iy >= 0) && (iy < 256) && (ix >= 0) && (ix < 256);
        const __bf16* row = in + (((size_t)b * 256 + iy) * 256 + ix) * 32;
        if (tap < 8) {   // prefetch next tap's row (L2-resident rereads)
            const int py = 2 * y - 1 + (tap + 1) / 3;
            const int px = 2 * x - 1 + (tap + 1) % 3;
            if (py >= 0 && py < 256 && px >= 0 && px < 256)
                __builtin_prefetch(in + (((size_t)b * 256 + py) * 256 + px) * 32, 0, 3);
        }
        v16bf A = load_A_seg(row, valid);
        const __bf16* wbase = wpk2 + tap * 4 * 512;
#pragma unroll
        for (int t = 0; t < 4; ++t) {
            v16bf B = load_B_packed(wbase + t * 512);
            acc[t]  = wmma_bf16(A, B, acc[t]);
        }
    }

    const size_t rowbase = ((size_t)b * 128 + y) * 128;
#pragma unroll
    for (int t = 0; t < 4; ++t) {
        const int   co = t * 16 + (l & 15);
        const float bb = bias[co];
#pragma unroll
        for (int r = 0; r < 8; ++r) {
            const int mm = r + ((l >> 4) << 3);
            float v = acc[t][r] + bb;
            v = v > 0.0f ? v : 0.0f;
            out[(rowbase + x0 + mm) * 64 + co] = (__bf16)v;
        }
    }
}

// ---------------------------------------------------------------------------
// conv3: 64->128, stride 2, pad 1. [B,128,128,64] -> z fp32 NCHW [B,128,64,64]
// (d_out) + bf16 NHWC copy [B,64,64,128] for correlation.
// Per wave: 16 px x 128 co; 9 taps x 2 K-chunks x 8 WMMA = 144 WMMA.
// ---------------------------------------------------------------------------
__global__ void conv3_wmma_kernel(const __bf16* __restrict__ in,
                                  const __bf16* __restrict__ wpk3,
                                  const float* __restrict__ bias,
                                  float* __restrict__ zout,
                                  __bf16* __restrict__ zbf) {
    const int b  = blockIdx.y;
    const int w  = (int)threadIdx.x >> 5;
    const int y  = blockIdx.x * 2 + (w >> 2); // 0..63
    const int x0 = (w & 3) * 16;
    const int l  = lane32();
    const int m  = l & 15;
    const int x  = x0 + m;

    v8f acc[8];
#pragma unroll
    for (int t = 0; t < 8; ++t)
#pragma unroll
        for (int r = 0; r < 8; ++r) acc[t][r] = 0.0f;

#pragma unroll
    for (int tap = 0; tap < 9; ++tap) {
        const int iy = 2 * y - 1 + tap / 3;
        const int ix = 2 * x - 1 + tap % 3;
        const bool valid = (iy >= 0) && (iy < 128) && (ix >= 0) && (ix < 128);
        const __bf16* row = in + (((size_t)b * 128 + iy) * 128 + ix) * 64;
        if (tap < 8) {
            const int py = 2 * y - 1 + (tap + 1) / 3;
            const int px = 2 * x - 1 + (tap + 1) % 3;
            if (py >= 0 && py < 128 && px >= 0 && px < 128)
                __builtin_prefetch(in + (((size_t)b * 128 + py) * 128 + px) * 64, 0, 3);
        }
#pragma unroll
        for (int kk = 0; kk < 2; ++kk) {
            v16bf A = load_A_seg(row + kk * 32, valid);
            const __bf16* wbase = wpk3 + (tap * 16 + kk * 8) * 512;
#pragma unroll
            for (int t = 0; t < 8; ++t) {
                v16bf B = load_B_packed(wbase + t * 512);
                acc[t]  = wmma_bf16(A, B, acc[t]);
            }
        }
    }

#pragma unroll
    for (int t = 0; t < 8; ++t) {
        const int   co = t * 16 + (l & 15);
        const float bb = bias[co];
#pragma unroll
        for (int r = 0; r < 8; ++r) {
            const int mm = r + ((l >> 4) << 3);
            const int xx = x0 + mm;
            float v = acc[t][r] + bb;
            v = v > 0.0f ? v : 0.0f;
            zout[(((size_t)b * 128 + co) * 64 + y) * 64 + xx] = v;           // NCHW f32
            zbf[(((size_t)b * 64 + y) * 64 + xx) * 128 + co]  = (__bf16)v;   // NHWC bf16
        }
    }
}

// ---------------------------------------------------------------------------
// Local correlation: corr[b, i*13+j, y, x] = <z_t[y,x,:], z_t1[y+i-6,x+j-6,:]>
// / sqrt(128). One thread per pixel; z_t row register-cached (b128 loads).
// Output-bandwidth-bound (88.6 MB writes); z_t1 rereads live in 192 MB L2.
// ---------------------------------------------------------------------------
__device__ __forceinline__ float bf_lo(unsigned u) {
    union { unsigned i; float f; } v; v.i = u << 16; return v.f;
}
__device__ __forceinline__ float bf_hi(unsigned u) {
    union { unsigned i; float f; } v; v.i = u & 0xffff0000u; return v.f;
}

__global__ void corr_kernel(const __bf16* __restrict__ zt,
                            const __bf16* __restrict__ zt1,
                            float* __restrict__ corr) {
    const int idx = blockIdx.x * 256 + (int)threadIdx.x;  // 32*64*64 threads
    const int b = idx >> 12;
    const int y = (idx >> 6) & 63;
    const int x = idx & 63;

    const v4u* rt = (const v4u*)(zt + (size_t)idx * 128);
    v4u row[16];
#pragma unroll
    for (int q = 0; q < 16; ++q) row[q] = rt[q];

    const float inv = 0.088388347648318447f;  // 1/sqrt(128)
    for (int i = 0; i < 13; ++i) {
        const int yy = y + i - 6;
        for (int j = 0; j < 13; ++j) {
            const int xx = x + j - 6;
            float s = 0.0f;
            if (yy >= 0 && yy < 64 && xx >= 0 && xx < 64) {
                const v4u* r1 = (const v4u*)(zt1 + (((size_t)b * 64 + yy) * 64 + xx) * 128);
                if (j < 12 && xx + 1 < 64)
                    __builtin_prefetch((const __bf16*)r1 + 128, 0, 3);
#pragma unroll
                for (int q = 0; q < 16; ++q) {
                    const v4u a = row[q];
                    const v4u c = r1[q];
#pragma unroll
                    for (int e = 0; e < 4; ++e) {
                        s += bf_lo(a[e]) * bf_lo(c[e]);
                        s += bf_hi(a[e]) * bf_hi(c[e]);
                    }
                }
            }
            corr[((size_t)(b * 169 + i * 13 + j) << 12) + y * 64 + x] = s * inv;
        }
    }
}

// ---------------------------------------------------------------------------
// h_t = corr.mean over HxW : one block per (b, offset), LDS tree reduce.
// ---------------------------------------------------------------------------
__global__ void reduce_mean_kernel(const float* __restrict__ corr,
                                   float* __restrict__ ht) {
    __shared__ float sm[256];
    const int bo = blockIdx.x;                       // 0..5407
    const float* p = corr + ((size_t)bo << 12);
    float s = 0.0f;
    for (int i = (int)threadIdx.x; i < 4096; i += 256) s += p[i];
    sm[threadIdx.x] = s;
    __syncthreads();
    for (int st = 128; st > 0; st >>= 1) {
        if ((int)threadIdx.x < st) sm[threadIdx.x] += sm[threadIdx.x + st];
        __syncthreads();
    }
    if (threadIdx.x == 0) ht[bo] = sm[0] * (1.0f / 4096.0f);
}

// ---------------------------------------------------------------------------
// pred = h_t @ w_head.T + b_head   ([32,169] x [169,2])
// ---------------------------------------------------------------------------
__global__ void head_kernel(const float* __restrict__ ht,
                            const float* __restrict__ wh,
                            const float* __restrict__ bh,
                            float* __restrict__ pred) {
    const int t = (int)threadIdx.x;
    if (t >= 64) return;
    const int b = t >> 1, j = t & 1;
    float s = bh[j];
    for (int k = 0; k < 169; ++k) s += ht[b * 169 + k] * wh[j * 169 + k];
    pred[t] = s;
}

// ---------------------------------------------------------------------------
// Launcher
// ---------------------------------------------------------------------------
extern "C" void kernel_launch(void* const* d_in, const int* in_sizes, int n_in,
                              void* d_out, int out_size, void* d_ws, size_t ws_size,
                              hipStream_t stream) {
    (void)in_sizes; (void)n_in; (void)out_size; (void)ws_size;

    const float* img_t  = (const float*)d_in[0];
    const float* img_t1 = (const float*)d_in[1];
    const float* w1 = (const float*)d_in[2];
    const float* b1 = (const float*)d_in[3];
    const float* w2 = (const float*)d_in[4];
    const float* b2 = (const float*)d_in[5];
    const float* w3 = (const float*)d_in[6];
    const float* b3 = (const float*)d_in[7];
    const float* w_head = (const float*)d_in[8];
    const float* b_head = (const float*)d_in[9];

    // d_out: pred[64] | z_t[32*128*64*64] | z_t1[...] | corr[32*169*64*64] | h_t[5408]
    float* out  = (float*)d_out;
    float* pred = out;
    float* z_t  = out + 64;
    float* z_t1 = z_t + (size_t)32 * 128 * 64 * 64;
    float* corr = z_t1 + (size_t)32 * 128 * 64 * 64;
    float* h_t  = corr + (size_t)32 * 169 * 64 * 64;

    // Workspace carve-up (bf16 activations, 256B aligned)
    char* ws = (char*)d_ws;
    auto carve = [&](size_t bytes) {
        char* p = ws;
        ws += (bytes + 255) & ~(size_t)255;
        return p;
    };
    __bf16* wpk1 = (__bf16*)carve((size_t)2 * 512 * 2);
    __bf16* wpk2 = (__bf16*)carve((size_t)36 * 512 * 2);
    __bf16* wpk3 = (__bf16*)carve((size_t)144 * 512 * 2);
    __bf16* buf1 = (__bf16*)carve((size_t)32 * 256 * 256 * 32 * 2);  // conv1 out
    __bf16* buf2 = (__bf16*)carve((size_t)32 * 128 * 128 * 64 * 2);  // conv2 out
    __bf16* zb_t  = (__bf16*)carve((size_t)32 * 64 * 64 * 128 * 2);
    __bf16* zb_t1 = (__bf16*)carve((size_t)32 * 64 * 64 * 128 * 2);

    pack_weights_kernel<<<364, 256, 0, stream>>>(w1, w2, w3, wpk1, wpk2, wpk3);

    for (int im = 0; im < 2; ++im) {
        const float* img = im ? img_t1 : img_t;
        float*  z  = im ? z_t1 : z_t;
        __bf16* zb = im ? zb_t1 : zb_t;
        conv1_wmma_kernel<<<dim3(2, 256, 32), 256, 0, stream>>>(img, wpk1, b1, buf1);
        conv2_wmma_kernel<<<dim3(128, 32), 256, 0, stream>>>(buf1, wpk2, b2, buf2);
        conv3_wmma_kernel<<<dim3(32, 32), 256, 0, stream>>>(buf2, wpk3, b3, z, zb);
    }

    corr_kernel<<<512, 256, 0, stream>>>(zb_t, zb_t1, corr);
    reduce_mean_kernel<<<32 * 169, 256, 0, stream>>>(corr, h_t);
    head_kernel<<<1, 64, 0, stream>>>(h_t, w_head, b_head, pred);
}
